// VectorQuantizer_57226144252164
// MI455X (gfx1250) — compile-verified, matrix operand order checked by reading, not run on hardware
//
#include <hip/hip_runtime.h>

#define NVEC 262144
#define DDIM 64
#define KCB  512

typedef __attribute__((ext_vector_type(16))) __bf16 v16bf;
typedef __attribute__((ext_vector_type(16))) short  v16s;
typedef __attribute__((ext_vector_type(8)))  float  v8f;
typedef __attribute__((ext_vector_type(4)))  unsigned int u32x4;

// 32-byte WMMA 16-bit operand: elements 0..15 in memory order; q[0]=elems 0-7, q[1]=elems 8-15
union Op16 {
  v16bf bf;
  v16s  s;
  u32x4 q[2];
};

__device__ __forceinline__ unsigned short f2bf(float f) {
  unsigned int u = __float_as_uint(f);
  u += 0x7FFFu + ((u >> 16) & 1u);   // round-to-nearest-even
  return (unsigned short)(u >> 16);
}
__device__ __forceinline__ float bf2f(unsigned short b) {
  return __uint_as_float(((unsigned int)b) << 16);
}

// Convert 8 contiguous fp32 into split-bf16 at element offset `off` of hi/lo operands
__device__ __forceinline__ void cv8(const float* __restrict__ p, Op16& hi, Op16& lo, int off) {
  float4 f0 = *(const float4*)(p);
  float4 f1 = *(const float4*)(p + 4);
  float v[8] = {f0.x, f0.y, f0.z, f0.w, f1.x, f1.y, f1.z, f1.w};
#pragma unroll
  for (int e = 0; e < 8; ++e) {
    unsigned short h = f2bf(v[e]);
    hi.s[off + e] = (short)h;
    lo.s[off + e] = (short)f2bf(v[e] - bf2f(h));
  }
}

// Fetch one 16-column B tile (split-bf16 of -2e) + ||e||^2 for this lane's column
__device__ __forceinline__ void loadBtile(const unsigned short* cb_hi,
                                          const unsigned short* cb_lo,
                                          const float* enormS,
                                          int col, int kbase,
                                          Op16* bh, Op16* bl, float& en) {
#pragma unroll
  for (int kc = 0; kc < 2; ++kc) {
    const unsigned short* ph = cb_hi + col * DDIM + kc * 32 + kbase;
    bh[kc].q[0] = *(const u32x4*)(ph);
    bh[kc].q[1] = *(const u32x4*)(ph + 16);
    const unsigned short* pl = cb_lo + col * DDIM + kc * 32 + kbase;
    bl[kc].q[0] = *(const u32x4*)(pl);
    bl[kc].q[1] = *(const u32x4*)(pl + 16);
  }
  en = enormS[col];
}

extern "C" __global__ __launch_bounds__(256)
void vq_main(const float* __restrict__ z, const float* __restrict__ cb,
             float* __restrict__ qout, float* __restrict__ idxout,
             float* __restrict__ wsum) {
  extern __shared__ char smem[];
  unsigned short* cb_hi  = (unsigned short*)smem;          // 512*64 bf16 hi of (-2e)
  unsigned short* cb_lo  = cb_hi + KCB * DDIM;             // 512*64 bf16 lo of (-2e)
  float*          enormS = (float*)(cb_lo + KCB * DDIM);   // 512 f32 ||e||^2
  __shared__ float wavesum[8];

  const int tid = threadIdx.x;

  // Codebook -> split-bf16 of (-2e) in LDS (folds the -2 of the distance form)
  for (int i = tid; i < KCB * DDIM; i += 256) {
    float e = -2.0f * cb[i];
    unsigned short h = f2bf(e);
    cb_hi[i] = h;
    cb_lo[i] = f2bf(e - bf2f(h));
  }
  // ||e_k||^2 (fp32 exact, from original codebook)
  for (int r = tid; r < KCB; r += 256) {
    const float4* row = (const float4*)(cb + r * DDIM);
    float s = 0.f;
#pragma unroll
    for (int i = 0; i < DDIM / 4; ++i) {
      float4 v = row[i];
      s += v.x * v.x + v.y * v.y + v.z * v.z + v.w * v.w;
    }
    enormS[r] = s;
  }
  __syncthreads();

  const int lane    = tid & 31;
  const int wave    = tid >> 5;
  const int rowBase = blockIdx.x * 256 + wave * 32;   // wave owns 32 rows
  const int lrow    = lane & 15;
  const int kbase   = (lane >= 16) ? 8 : 0;           // 16-bit operand K striping

  // A tiles (z rows) as split-bf16: [2 row tiles][2 K-chunks of 32]
  Op16 a_hi[2][2], a_lo[2][2];
#pragma unroll
  for (int rt = 0; rt < 2; ++rt) {
    const float* zr = z + (size_t)(rowBase + rt * 16 + lrow) * DDIM;
#pragma unroll
    for (int kc = 0; kc < 2; ++kc) {
      cv8(zr + kc * 32 + kbase,      a_hi[rt][kc], a_lo[rt][kc], 0);
      cv8(zr + kc * 32 + 16 + kbase, a_hi[rt][kc], a_lo[rt][kc], 8);
    }
  }

  float minv[2][8];
  int   mini[2][8];
#pragma unroll
  for (int rt = 0; rt < 2; ++rt)
#pragma unroll
    for (int j = 0; j < 8; ++j) { minv[rt][j] = 3.4e38f; mini[rt][j] = 0; }

  // Double-buffered B tiles: prefetch t+1 while computing t
  Op16 bh[2][2], bl[2][2];
  float en[2];
  loadBtile(cb_hi, cb_lo, enormS, lrow, kbase, bh[0], bl[0], en[0]);

#pragma unroll 2
  for (int t = 0; t < KCB / 16; ++t) {
    const int s  = t & 1;
    const int tn = (t + 1) & (KCB / 16 - 1);    // wraps to 0 on last iter (harmless)
    loadBtile(cb_hi, cb_lo, enormS, tn * 16 + lrow, kbase, bh[s ^ 1], bl[s ^ 1], en[s ^ 1]);

    const int col = t * 16 + lrow;
#pragma unroll
    for (int rt = 0; rt < 2; ++rt) {
      v8f acc;
#pragma unroll
      for (int j = 0; j < 8; ++j) acc[j] = en[s];   // seed with ||e||^2
#pragma unroll
      for (int kc = 0; kc < 2; ++kc) {
        acc = __builtin_amdgcn_wmma_f32_16x16x32_bf16(false, a_hi[rt][kc].bf, false, bh[s][kc].bf, (short)0, acc, false, false);
        acc = __builtin_amdgcn_wmma_f32_16x16x32_bf16(false, a_lo[rt][kc].bf, false, bh[s][kc].bf, (short)0, acc, false, false);
        acc = __builtin_amdgcn_wmma_f32_16x16x32_bf16(false, a_hi[rt][kc].bf, false, bl[s][kc].bf, (short)0, acc, false, false);
      }
      // acc == ||e||^2 - 2 z.e  (row-constant ||z||^2 dropped)
#pragma unroll
      for (int j = 0; j < 8; ++j) {
        if (acc[j] < minv[rt][j]) { minv[rt][j] = acc[j]; mini[rt][j] = col; }
      }
    }
  }

  // Cross-lane argmin (within each 16-lane half; C layout: lanes0-15 rows j, lanes16-31 rows j+8)
  float lsum = 0.0f;
#pragma unroll
  for (int rt = 0; rt < 2; ++rt) {
#pragma unroll
    for (int j = 0; j < 8; ++j) {
#pragma unroll
      for (int m = 8; m >= 1; m >>= 1) {
        float ov = __shfl_xor(minv[rt][j], m, 32);
        int   oi = __shfl_xor(mini[rt][j], m, 32);
        if (ov < minv[rt][j] || (ov == minv[rt][j] && oi < mini[rt][j])) {
          minv[rt][j] = ov; mini[rt][j] = oi;
        }
      }
    }
    // lane pairs: lane l -> row r = l>>1 of this tile, half (l&1) of 64 dims
    const int r    = lane >> 1;
    const int idx  = mini[rt][r & 7];
    const int grow = rowBase + rt * 16 + r;
    const int d0   = (lane & 1) * 32;
    const float4* qsrc = (const float4*)(cb + (size_t)idx * DDIM + d0);
    const float4* zsrc = (const float4*)(z + (size_t)grow * DDIM + d0);
    float4* qdst = (float4*)(qout + (size_t)grow * DDIM + d0);
#pragma unroll
    for (int i = 0; i < 8; ++i) {
      float4 qv = qsrc[i];
      float4 zv = zsrc[i];
      qdst[i] = qv;
      float dx = qv.x - zv.x, dy = qv.y - zv.y, dz = qv.z - zv.z, dw = qv.w - zv.w;
      lsum += dx * dx + dy * dy + dz * dz + dw * dw;
    }
    if ((lane & 1) == 0) idxout[grow] = (float)idx;
  }

  // Deterministic block partial: wave butterfly -> fixed-order sum by thread 0
#pragma unroll
  for (int m = 16; m >= 1; m >>= 1) lsum += __shfl_xor(lsum, m, 32);
  if (lane == 0) wavesum[wave] = lsum;
  __syncthreads();
  if (tid == 0) {
    float s = 0.f;
#pragma unroll
    for (int w = 0; w < 8; ++w) s += wavesum[w];
    wsum[blockIdx.x] = s;
  }
}

extern "C" __global__ void vq_loss_finalize(const float* __restrict__ wsum,
                                            float* __restrict__ lossOut, int nblk) {
  const int lane = threadIdx.x;  // 32 threads
  float s = 0.f;
  const int chunk = nblk / 32;
  for (int i = 0; i < chunk; ++i) s += wsum[lane * chunk + i];
#pragma unroll
  for (int m = 16; m >= 1; m >>= 1) s += __shfl_xor(s, m, 32);
  if (lane == 0)
    *lossOut = 1.25f * s / (float)((long long)NVEC * DDIM);  // codebook + 0.25*commit
}

extern "C" void kernel_launch(void* const* d_in, const int* in_sizes, int n_in,
                              void* d_out, int out_size, void* d_ws, size_t ws_size,
                              hipStream_t stream) {
  const float* z  = (const float*)d_in[0];
  const float* cb = (const float*)d_in[1];
  float* out     = (float*)d_out;
  float* qout    = out;                            // [N*D] q_st
  float* lossOut = out + (size_t)NVEC * DDIM;      // [1] loss
  float* idxout  = lossOut + 1;                    // [N] idx (as float)
  float* wsum    = (float*)d_ws;                   // [nblk] partial sq-sums

  const int nblk = NVEC / 256;                     // 1024 blocks, 256 rows each
  const size_t shbytes = (size_t)KCB * DDIM * 2 * sizeof(unsigned short)
                       + (size_t)KCB * sizeof(float);  // 133120 B LDS
  vq_main<<<nblk, 256, shbytes, stream>>>(z, cb, qout, idxout, wsum);
  vq_loss_finalize<<<1, 32, 0, stream>>>(wsum, lossOut, nblk);
}
